// GNN_MultipleOutput_39702677684847
// MI455X (gfx1250) — matrix operand order repaired and checked by reading, not compiled, not run on hardware
//
#include <hip/hip_runtime.h>
#include <hip/hip_bf16.h>

#define N_NODES 50000
#define N_EDGES 625000
#define DF      128
#define NODE_F  (N_NODES * DF)      // 6,400,000

typedef __attribute__((ext_vector_type(16))) __bf16 v16bf;
typedef __attribute__((ext_vector_type(8)))  __bf16 v8bf;
typedef __attribute__((ext_vector_type(8)))  float  v8f;

// ---------- helpers ----------
__device__ __forceinline__ __bf16 to_bf16(float f) {
  // round-to-nearest-even f32 -> bf16
  unsigned u = __float_as_uint(f);
  unsigned r = (u + 0x7FFFu + ((u >> 16) & 1u)) >> 16;
  unsigned short s = (unsigned short)r;
  return *reinterpret_cast<__bf16*>(&s);
}
__device__ __forceinline__ float bf16_to_f32(unsigned short s) {
  return __uint_as_float(((unsigned)s) << 16);
}

// ---------- utility kernels ----------
__global__ void zero4_kernel(float4* p, int n4) {
  int i = blockIdx.x * blockDim.x + threadIdx.x;
  if (i < n4) p[i] = make_float4(0.f, 0.f, 0.f, 0.f);
}

__global__ void degree_kernel(const long long* __restrict__ ei, float* __restrict__ deg) {
  int e = blockIdx.x * blockDim.x + threadIdx.x;
  if (e >= N_EDGES) return;
  int dst = (int)ei[N_EDGES + e];
  atomicAdd(&deg[dst], 1.0f);
}

// one thread handles 4 floats of one edge's feature row (32 threads / edge)
__global__ void scatter_add_f32(const float* __restrict__ feat,
                                const long long* __restrict__ ei,
                                float* __restrict__ agg) {
  int tid = blockIdx.x * blockDim.x + threadIdx.x;   // < 20,000,000
  int e = tid >> 5;
  int k = (tid & 31) << 2;
  int src = (int)ei[e];
  int dst = (int)ei[N_EDGES + e];
  const float4 v = *(const float4*)(feat + (size_t)src * DF + k);
  float* dp = agg + (size_t)dst * DF + k;
  atomicAdd(dp + 0, v.x);
  atomicAdd(dp + 1, v.y);
  atomicAdd(dp + 2, v.z);
  atomicAdd(dp + 3, v.w);
}

__global__ void scatter_add_bf16(const unsigned short* __restrict__ feat,
                                 const long long* __restrict__ ei,
                                 float* __restrict__ agg) {
  int tid = blockIdx.x * blockDim.x + threadIdx.x;
  int e = tid >> 5;
  int k = (tid & 31) << 2;
  int src = (int)ei[e];
  int dst = (int)ei[N_EDGES + e];
  const ushort4 v = *(const ushort4*)(feat + (size_t)src * DF + k);
  float* dp = agg + (size_t)dst * DF + k;
  atomicAdd(dp + 0, bf16_to_f32(v.x));
  atomicAdd(dp + 1, bf16_to_f32(v.y));
  atomicAdd(dp + 2, bf16_to_f32(v.z));
  atomicAdd(dp + 3, bf16_to_f32(v.w));
}

__global__ void cvt_f32_bf16(const float* __restrict__ in, __bf16* __restrict__ out, int n) {
  int i = blockIdx.x * blockDim.x + threadIdx.x;
  if (i < n) out[i] = to_bf16(in[i]);
}

// mean = agg / max(deg,1), converted to bf16
__global__ void meancvt_kernel(const float* __restrict__ agg,
                               const float* __restrict__ deg,
                               __bf16* __restrict__ out) {
  int i = blockIdx.x * blockDim.x + threadIdx.x;
  if (i >= NODE_F) return;
  int node = i >> 7;
  float d = fmaxf(deg[node], 1.0f);
  out[i] = to_bf16(agg[i] / d);
}

// ---------- WMMA GEMM core ----------
// Computes a 16(node) x 16(out) f32 tile:  A1 @ W1^T + A2 @ W2^T
// A* : [N_NODES x 128] bf16 row-major activations
// W* : [128 x 128] bf16 row-major weights (out-major) -> rows ARE B columns
__device__ __forceinline__ v8f sage_tile(const __bf16* __restrict__ A1,
                                         const __bf16* __restrict__ W1,
                                         const __bf16* __restrict__ A2,
                                         const __bf16* __restrict__ W2,
                                         int nbase, int obase, int lane) {
  const int half = lane >> 4;    // 0: lanes 0-15, 1: lanes 16-31
  const int lr   = lane & 15;
  const __bf16* a1 = A1 + (size_t)(nbase + lr) * DF;
  const __bf16* a2 = A2 + (size_t)(nbase + lr) * DF;
  const __bf16* w1 = W1 + (size_t)(obase + lr) * DF;
  const __bf16* w2 = W2 + (size_t)(obase + lr) * DF;

  v8f acc = {};
#pragma unroll
  for (int kt = 0; kt < 4; ++kt) {
    const int k0 = kt * 32;
    union { v16bf v; v8bf h[2]; } af, bf_;
    // A fragment (16x32 bf16): lane half=0 -> K[k0..k0+7] & K[k0+16..k0+23]
    //                          lane half=1 -> K[k0+8..k0+15] & K[k0+24..k0+31]
    af.h[0] = *(const v8bf*)(a1 + k0 + half * 8);
    af.h[1] = *(const v8bf*)(a1 + k0 + 16 + half * 8);
    // B fragment (32x16 bf16): col n = lane&15 (== weight row), same K split as A
    bf_.h[0] = *(const v8bf*)(w1 + k0 + half * 8);
    bf_.h[1] = *(const v8bf*)(w1 + k0 + 16 + half * 8);
    acc = __builtin_amdgcn_wmma_f32_16x16x32_bf16(false, af.v, false, bf_.v,
                                                  (short)0, acc, false, false);
    af.h[0] = *(const v8bf*)(a2 + k0 + half * 8);
    af.h[1] = *(const v8bf*)(a2 + k0 + 16 + half * 8);
    bf_.h[0] = *(const v8bf*)(w2 + k0 + half * 8);
    bf_.h[1] = *(const v8bf*)(w2 + k0 + 16 + half * 8);
    acc = __builtin_amdgcn_wmma_f32_16x16x32_bf16(false, af.v, false, bf_.v,
                                                  (short)0, acc, false, false);
  }
  return acc;
}

// Layer 1: h = relu(mean@Wl^T + b + x@Wr^T) -> bf16
__global__ void __launch_bounds__(256)
gemm1_kernel(const __bf16* __restrict__ meanbf, const __bf16* __restrict__ wl,
             const __bf16* __restrict__ xbf,    const __bf16* __restrict__ wr,
             const float* __restrict__ bias, __bf16* __restrict__ hout) {
  const int lane  = threadIdx.x & 31;
  const int wave  = threadIdx.x >> 5;
  const int nbase = blockIdx.x * 16;
  const int obase = wave * 16;
  v8f acc = sage_tile(meanbf, wl, xbf, wr, nbase, obase, lane);
  const int half = lane >> 4, lr = lane & 15;
  const int o = obase + lr;
  const float bv = bias[o];
#pragma unroll
  for (int v = 0; v < 8; ++v) {
    const int node = nbase + v + 8 * half;   // D row m = v + 8*(lane/16)
    float val = acc[v] + bv;
    val = fmaxf(val, 0.0f);
    hout[(size_t)node * DF + o] = to_bf16(val);
  }
}

// Layer 2: y = mean@Wl^T + b + h@Wr^T -> f32, replicated 4x into d_out
__global__ void __launch_bounds__(256)
gemm2_kernel(const __bf16* __restrict__ meanbf, const __bf16* __restrict__ wl,
             const __bf16* __restrict__ hbf,    const __bf16* __restrict__ wr,
             const float* __restrict__ bias, float* __restrict__ out) {
  const int lane  = threadIdx.x & 31;
  const int wave  = threadIdx.x >> 5;
  const int nbase = blockIdx.x * 16;
  const int obase = wave * 16;
  v8f acc = sage_tile(meanbf, wl, hbf, wr, nbase, obase, lane);
  const int half = lane >> 4, lr = lane & 15;
  const int o = obase + lr;
  const float bv = bias[o];
#pragma unroll
  for (int v = 0; v < 8; ++v) {
    const int node = nbase + v + 8 * half;
    const float val = acc[v] + bv;
    const size_t idx = (size_t)node * DF + o;
    out[idx]              = val;
    out[idx + 1 * NODE_F] = val;
    out[idx + 2 * NODE_F] = val;
    out[idx + 3 * NODE_F] = val;
  }
}

// ---------- launcher ----------
extern "C" void kernel_launch(void* const* d_in, const int* in_sizes, int n_in,
                              void* d_out, int out_size, void* d_ws, size_t ws_size,
                              hipStream_t stream) {
  const float*     x   = (const float*)d_in[0];
  const long long* ei  = (const long long*)d_in[1];   // int64 edge_index [2, E]
  const float*     W1l = (const float*)d_in[2];
  const float*     b1  = (const float*)d_in[3];
  const float*     W1r = (const float*)d_in[4];
  const float*     W2l = (const float*)d_in[5];
  const float*     b2  = (const float*)d_in[6];
  const float*     W2r = (const float*)d_in[7];

  char* ws = (char*)d_ws;
  float*  deg = (float*)(ws);                 //   200,000 B
  float*  agg = (float*)(ws +    200000);     // 25,600,000 B
  __bf16* xbf = (__bf16*)(ws + 25800000);     // 12,800,000 B
  __bf16* hbf = (__bf16*)(ws + 38600000);     // 12,800,000 B
  __bf16* mbf = (__bf16*)(ws + 51400000);     // 12,800,000 B
  __bf16* w1l = (__bf16*)(ws + 64200000);     //     32,768 B each
  __bf16* w1r = (__bf16*)(ws + 64232768);
  __bf16* w2l = (__bf16*)(ws + 64265536);
  __bf16* w2r = (__bf16*)(ws + 64298304);     // total ~64.4 MB

  const int WN = DF * DF;                     // 16384
  const int scatterThreads = N_EDGES * 32;    // 20,000,000

  // zero deg + agg in one pass (adjacent): 6,450,000 floats
  {
    int n4 = (N_NODES + NODE_F) / 4;
    zero4_kernel<<<(n4 + 255) / 256, 256, 0, stream>>>((float4*)ws, n4);
  }
  degree_kernel<<<(N_EDGES + 255) / 256, 256, 0, stream>>>(ei, deg);
  scatter_add_f32<<<scatterThreads / 256, 256, 0, stream>>>(x, ei, agg);

  cvt_f32_bf16<<<(NODE_F + 255) / 256, 256, 0, stream>>>(x, xbf, NODE_F);
  cvt_f32_bf16<<<(WN + 255) / 256, 256, 0, stream>>>(W1l, w1l, WN);
  cvt_f32_bf16<<<(WN + 255) / 256, 256, 0, stream>>>(W1r, w1r, WN);
  cvt_f32_bf16<<<(WN + 255) / 256, 256, 0, stream>>>(W2l, w2l, WN);
  cvt_f32_bf16<<<(WN + 255) / 256, 256, 0, stream>>>(W2r, w2r, WN);

  meancvt_kernel<<<(NODE_F + 255) / 256, 256, 0, stream>>>(agg, deg, mbf);
  gemm1_kernel<<<N_NODES / 16, 256, 0, stream>>>(mbf, w1l, xbf, w1r, b1, hbf);

  // layer 2 aggregation over h
  {
    int n4 = NODE_F / 4;
    zero4_kernel<<<(n4 + 255) / 256, 256, 0, stream>>>((float4*)agg, n4);
  }
  scatter_add_bf16<<<scatterThreads / 256, 256, 0, stream>>>((const unsigned short*)hbf, ei, agg);
  meancvt_kernel<<<(NODE_F + 255) / 256, 256, 0, stream>>>(agg, deg, mbf);
  gemm2_kernel<<<N_NODES / 16, 256, 0, stream>>>(mbf, w2l, hbf, w2r, b2, (float*)d_out);
}
